// ResidualVectorQuantizer_79697413144711
// MI455X (gfx1250) — matrix operand order repaired and checked by reading, not compiled, not run on hardware
//
#include <hip/hip_runtime.h>

// ---------------------------------------------------------------------------
// Residual VQ encode for MI455X (gfx1250, wave32, WMMA).
//   hidden [N,256] f32, codebooks [8,1024,256] f32
//   -> codes [8,N] (stored as float), quantized [N,256] f32
// argmin_k (c_sq[k] - 2 r.c_k) per level via bf16 WMMA, f32 accumulate.
// Residual lives in LDS f32; quantized = hidden - residual_final.
// Codebook tiles staged fragment-major in LDS (conflict-free B reads),
// double-buffered via async global->LDS loads (ASYNCcnt) on gfx1250.
// ---------------------------------------------------------------------------

typedef __attribute__((ext_vector_type(16))) __bf16 v16bf;
typedef __attribute__((ext_vector_type(8)))  float  v8f;

#define D_DIM 256
#define K_CB  1024
#define NQ    8
#define ROWS_PER_BLOCK 128
#define ROWS_PER_WAVE  16
#define RP 264                 // padded residual row (floats)
#define TN 128                 // codes per stage tile

#define RES_BYTES (ROWS_PER_BLOCK * RP * 4)          // 135168
#define CBT_BYTES (TN * D_DIM * 2)                   // 65536 (fragment-major, no pad)
#define CSQ_BYTES (K_CB * 4)                         // 4096
#define SMEM_BYTES (RES_BYTES + 2 * CBT_BYTES + CSQ_BYTES)  // 270336

#define USE_ASYNC_COPY 1

__device__ __forceinline__ void wait_async_zero()
{
#if defined(__gfx1250__) && USE_ASYNC_COPY
    asm volatile("s_wait_asynccnt 0x0" ::: "memory");
#endif
}

// Stage TN codes x 256 bf16 into a fragment-major LDS tile:
//   fragment block (g, c) = 1KB at ((g*8+c)<<10); lane l's 32B at lane*32.
//   Element (code, k): g=code/16, n=code%16, c=k/32, H'=(k%32)/16, j=k%16
//   -> byte off = ((g*8+c)<<10) + ((n+16H')<<5) + 2j
__device__ __forceinline__ void issue_stage(char* __restrict__ dst_tile,
                                            const __bf16* __restrict__ cbbf,
                                            int q, int st, int tid)
{
    const char* gbase =
        (const char*)(cbbf + ((size_t)q * K_CB + (size_t)st * TN) * D_DIM);
    #pragma unroll
    for (int p = 0; p < 16; p++) {
        const int e    = p * 256 + tid;     // 16B chunk id
        const int code = e >> 5;            // 32 chunks per code row
        const int k0   = (e & 31) << 3;     // element offset within row
        const int g  = code >> 4, n = code & 15;
        const int c  = k0 >> 5, Hp = (k0 >> 4) & 1, j0 = k0 & 15;
        const int ldsoff = ((g * 8 + c) << 10) + ((n + (Hp << 4)) << 5) + (j0 << 1);
        const char* gsrc = gbase + ((size_t)code * D_DIM + k0) * 2;
#if defined(__gfx1250__) && USE_ASYNC_COPY
        // Async DMA into LDS, tracked by ASYNCcnt. VDST VGPR = LDS byte
        // address (low 32 bits of a generic LDS pointer; dynamic LDS starts
        // at offset 0 here). VADDR = 64-bit global address, no SADDR.
        asm volatile("global_load_async_to_lds_b128 %0, %1, off"
                     :: "v"((unsigned)(unsigned long long)(dst_tile + ldsoff)),
                        "v"(gsrc)
                     : "memory");
#else
        *(uint4*)(dst_tile + ldsoff) = *(const uint4*)gsrc;
#endif
    }
}

// ---------------------------------------------------------------------------
// Prepass: codebooks f32 -> bf16, plus ||c||^2. One block per codebook row.
// ---------------------------------------------------------------------------
extern "C" __global__ __launch_bounds__(256) void rvq_prep(
    const float* __restrict__ cb, __bf16* __restrict__ cbbf,
    float* __restrict__ csq)
{
    __shared__ float s[256];
    const int row = blockIdx.x;          // q*K + k
    const int d   = threadIdx.x;
    float v = cb[(size_t)row * D_DIM + d];
    cbbf[(size_t)row * D_DIM + d] = (__bf16)v;
    s[d] = v * v;
    __syncthreads();
    #pragma unroll
    for (int off = 128; off > 0; off >>= 1) {
        if (d < off) s[d] += s[d + off];
        __syncthreads();
    }
    if (d == 0) csq[row] = s[0];
}

// ---------------------------------------------------------------------------
// Main kernel: 256 threads (8 waves). Each wave owns 16 rows; all 8 RVQ
// levels in one pass with residual resident in LDS.
// ---------------------------------------------------------------------------
extern "C" __global__ __launch_bounds__(256) void rvq_main(
    const float* __restrict__ hidden, const float* __restrict__ cb,
    const __bf16* __restrict__ cbbf, const float* __restrict__ csqg,
    float* __restrict__ out_codes, float* __restrict__ out_quant, int N)
{
    extern __shared__ char smem[];
    float* resid = (float*)smem;                             // [128][RP]
    char*  cbt0  = smem + RES_BYTES;                         // tile buf 0
    char*  cbt1  = smem + RES_BYTES + CBT_BYTES;             // tile buf 1
    float* csq   = (float*)(smem + RES_BYTES + 2 * CBT_BYTES);

    const int tid  = threadIdx.x;
    const int lane = tid & 31;
    const int w    = tid >> 5;
    const int hh   = lane >> 4;
    const int l16  = lane & 15;
    const int rowBase = blockIdx.x * ROWS_PER_BLOCK;

    // ---- hidden rows -> LDS residual
    {
        const int r  = tid >> 1;
        const int h2 = tid & 1;
        const float4* src = (const float4*)(hidden + (size_t)(rowBase + r) * D_DIM + h2 * 128);
        float4* dst = (float4*)(resid + r * RP + h2 * 128);
        #pragma unroll
        for (int i = 0; i < 32; i++) dst[i] = src[i];
    }
    __syncthreads();

    const int myrowA = w * ROWS_PER_WAVE + l16;

    #pragma unroll 1
    for (int q = 0; q < NQ; q++) {
        // ---- stage c_sq
        #pragma unroll
        for (int i = 0; i < 4; i++)
            csq[tid + i * 256] = csqg[q * K_CB + tid + i * 256];

        // ---- A fragments (16x32 bf16; lanes0-15: K 0-15, lanes16-31: 16-31)
        v16bf a[8];
        {
            const float* rp = resid + myrowA * RP;
            #pragma unroll
            for (int c = 0; c < 8; c++) {
                const int base = c * 32 + hh * 16;
                v16bf t;
                #pragma unroll
                for (int j = 0; j < 8; j++) t[j]     = (__bf16)rp[base + j];
                #pragma unroll
                for (int j = 0; j < 8; j++) t[8 + j] = (__bf16)rp[base + 8 + j];
                a[c] = t;
            }
        }

        float mv0,mv1,mv2,mv3,mv4,mv5,mv6,mv7;
        int   mi0,mi1,mi2,mi3,mi4,mi5,mi6,mi7;
        mv0=mv1=mv2=mv3=mv4=mv5=mv6=mv7 = 3.4e38f;
        mi0=mi1=mi2=mi3=mi4=mi5=mi6=mi7 = 0;

        // ---- prologue: stage 0
        issue_stage(cbt0, cbbf, q, 0, tid);
        wait_async_zero();
        __syncthreads();

        #pragma unroll 1
        for (int st = 0; st < K_CB / TN; st++) {
            char* cur = (st & 1) ? cbt1 : cbt0;
            if (st < K_CB / TN - 1)
                issue_stage((st & 1) ? cbt0 : cbt1, cbbf, q, st + 1, tid);

            // ---- compute: dual accumulator chains over two code groups
            #pragma unroll 1
            for (int gp = 0; gp < TN / 32; gp++) {
                v8f acc0 = {0.f,0.f,0.f,0.f,0.f,0.f,0.f,0.f};
                v8f acc1 = {0.f,0.f,0.f,0.f,0.f,0.f,0.f,0.f};
                const char* b0p = cur + ((gp * 2) * 8) * 1024 + lane * 32;
                const char* b1p = b0p + 8 * 1024;
                #pragma unroll
                for (int c = 0; c < 8; c++) {
                    v16bf b0 = *(const v16bf*)(b0p + c * 1024);
                    v16bf b1 = *(const v16bf*)(b1p + c * 1024);
                    acc0 = __builtin_amdgcn_wmma_f32_16x16x32_bf16(
                        false, a[c], false, b0, (short)0, acc0, false, false);
                    acc1 = __builtin_amdgcn_wmma_f32_16x16x32_bf16(
                        false, a[c], false, b1, (short)0, acc1, false, false);
                }
                const int code0 = st * TN + gp * 32 + l16;
                const int code1 = code0 + 16;
                const float cs0 = csq[code0];
                const float cs1 = csq[code1];
                float s;
                s = cs0 - 2.0f*acc0[0]; if (s < mv0){mv0=s; mi0=code0;}
                s = cs0 - 2.0f*acc0[1]; if (s < mv1){mv1=s; mi1=code0;}
                s = cs0 - 2.0f*acc0[2]; if (s < mv2){mv2=s; mi2=code0;}
                s = cs0 - 2.0f*acc0[3]; if (s < mv3){mv3=s; mi3=code0;}
                s = cs0 - 2.0f*acc0[4]; if (s < mv4){mv4=s; mi4=code0;}
                s = cs0 - 2.0f*acc0[5]; if (s < mv5){mv5=s; mi5=code0;}
                s = cs0 - 2.0f*acc0[6]; if (s < mv6){mv6=s; mi6=code0;}
                s = cs0 - 2.0f*acc0[7]; if (s < mv7){mv7=s; mi7=code0;}
                s = cs1 - 2.0f*acc1[0]; if (s < mv0){mv0=s; mi0=code1;}
                s = cs1 - 2.0f*acc1[1]; if (s < mv1){mv1=s; mi1=code1;}
                s = cs1 - 2.0f*acc1[2]; if (s < mv2){mv2=s; mi2=code1;}
                s = cs1 - 2.0f*acc1[3]; if (s < mv3){mv3=s; mi3=code1;}
                s = cs1 - 2.0f*acc1[4]; if (s < mv4){mv4=s; mi4=code1;}
                s = cs1 - 2.0f*acc1[5]; if (s < mv5){mv5=s; mi5=code1;}
                s = cs1 - 2.0f*acc1[6]; if (s < mv6){mv6=s; mi6=code1;}
                s = cs1 - 2.0f*acc1[7]; if (s < mv7){mv7=s; mi7=code1;}
            }
            wait_async_zero();     // next tile landed (per-wave), publish at barrier
            __syncthreads();
        }

        // ---- per-row argmin across each 16-lane half, then residual update
        float mvs[8] = {mv0,mv1,mv2,mv3,mv4,mv5,mv6,mv7};
        int   mis[8] = {mi0,mi1,mi2,mi3,mi4,mi5,mi6,mi7};
        #pragma unroll
        for (int j = 0; j < 8; j++) {
            float mv = mvs[j]; int mi = mis[j];
            #pragma unroll
            for (int off = 8; off >= 1; off >>= 1) {
                float ov = __shfl_xor(mv, off, 16);
                int   oi = __shfl_xor(mi, off, 16);
                if (ov < mv || (ov == mv && oi < mi)) { mv = ov; mi = oi; }
            }
            const int rloc = w * ROWS_PER_WAVE + j + 8 * hh;   // lanes0-15: row j; 16-31: j+8
            if (l16 == 0)
                out_codes[(size_t)q * N + rowBase + rloc] = (float)mi;
            const float* crow = cb + ((size_t)q * K_CB + mi) * D_DIM;
            float* rr = resid + rloc * RP;
            #pragma unroll
            for (int t2 = 0; t2 < 16; t2++) {
                const int d = l16 + 16 * t2;
                rr[d] -= crow[d];
            }
        }
    }

    __syncthreads();
    // ---- quantized = hidden - residual_final
    {
        const int r  = tid >> 1;
        const int h2 = tid & 1;
        const float4* hsrc = (const float4*)(hidden + (size_t)(rowBase + r) * D_DIM + h2 * 128);
        const float4* rsrc = (const float4*)(resid + r * RP + h2 * 128);
        float4* qdst = (float4*)(out_quant + (size_t)(rowBase + r) * D_DIM + h2 * 128);
        #pragma unroll
        for (int i = 0; i < 32; i++) {
            float4 hv = hsrc[i], rv = rsrc[i];
            float4 o; o.x=hv.x-rv.x; o.y=hv.y-rv.y; o.z=hv.z-rv.z; o.w=hv.w-rv.w;
            qdst[i] = o;
        }
    }
}

// ---------------------------------------------------------------------------
extern "C" void kernel_launch(void* const* d_in, const int* in_sizes, int n_in,
                              void* d_out, int out_size, void* d_ws, size_t ws_size,
                              hipStream_t stream)
{
    const float* hidden = (const float*)d_in[0];
    const float* cb     = (const float*)d_in[1];
    const int N = in_sizes[0] / D_DIM;          // 65536

    __bf16* cbbf = (__bf16*)d_ws;                                   // 4 MB
    float*  csq  = (float*)((char*)d_ws + (size_t)NQ * K_CB * D_DIM * sizeof(__bf16));

    float* out_codes = (float*)d_out;                    // [NQ, N] as float
    float* out_quant = out_codes + (size_t)NQ * N;       // [N, D]

    rvq_prep<<<NQ * K_CB, 256, 0, stream>>>(cb, cbbf, csq);

    (void)hipFuncSetAttribute((const void*)rvq_main,
                              hipFuncAttributeMaxDynamicSharedMemorySize,
                              (int)SMEM_BYTES);
    rvq_main<<<N / ROWS_PER_BLOCK, 256, SMEM_BYTES, stream>>>(
        hidden, cb, cbbf, csq, out_codes, out_quant, N);
}
